// A3TGCN2_76149770158682
// MI455X (gfx1250) — matrix operand order repaired and checked by reading, not compiled, not run on hardware
//
#include <hip/hip_runtime.h>
#include <hip/hip_bf16.h>
#include <math.h>

// ---------------------------------------------------------------------------
// A3TGCN2 with H0 == 0:  R-gate is dead; per-timestep cell reduces to
//   out_t = sigmoid(-(Y_t @ Mz^T + cz)) * tanh(Y_t @ Mh^T + ch)
//   result = sum_t softmax(att)_t * out_t
// where Y = A_norm @ X  (propagate raw 256-float feature block once),
//   Mz = Lz[:, :32] @ Wz,  cz = Lz[:, :32] @ bz + lz_b   (same for h).
// Dense gate phase uses v_wmma_f32_16x16x32_f16 (K=32 == F_IN), inline-0 C,
// bias folded into a branch-free exp2/rcp epilogue.
// ---------------------------------------------------------------------------

#define F_IN 32
#define T_STEPS 8
#define O_CH 32
#define FT 256          // F_IN * T_STEPS floats per node
#define LOG2E 1.4426950408889634f

typedef __attribute__((ext_vector_type(16))) _Float16 v16h;
typedef __attribute__((ext_vector_type(8)))  float    v8f;

// ---- Phase 1: degree / norm ------------------------------------------------
__global__ void k_deg_init(float* __restrict__ deg, int n) {
    int i = blockIdx.x * blockDim.x + threadIdx.x;
    if (i < n) deg[i] = 1.0f;                      // self-loop weight
}

__global__ void k_deg_accum(const long long* __restrict__ ei,
                            const float* __restrict__ ew,
                            float* __restrict__ deg, int e) {
    int i = blockIdx.x * blockDim.x + threadIdx.x;
    if (i < e) atomicAdd(&deg[(int)ei[(size_t)e + i]], ew[i]);
}

__global__ void k_dinv(float* __restrict__ deg, int n) {
    int i = blockIdx.x * blockDim.x + threadIdx.x;
    if (i < n) deg[i] = rsqrtf(deg[i]);            // deg >= 1 always
}

// ---- Phase 2: Y = A_norm @ X  (256 floats per node) ------------------------
__global__ void k_y_init(const float4* __restrict__ X4,
                         const float* __restrict__ dinv,
                         float4* __restrict__ Y4, int n) {
    int i = blockIdx.x * blockDim.x + threadIdx.x;   // over n*64 float4s
    if (i < n * (FT / 4)) {
        float d = dinv[i >> 6];
        float s = d * d;                              // self-loop norm
        float4 x = X4[i];
        Y4[i] = make_float4(s * x.x, s * x.y, s * x.z, s * x.w);
    }
}

__global__ void k_edge_prop(const float* __restrict__ X,
                            const long long* __restrict__ ei,
                            const float* __restrict__ ew,
                            const float* __restrict__ dinv,
                            float* __restrict__ Y, int e) {
    int wid  = (blockIdx.x * blockDim.x + threadIdx.x) >> 5;   // 1 wave / edge
    int lane = threadIdx.x & 31;
    if (wid >= e) return;                                       // wave-uniform
    int r = (int)ei[wid];
    int c = (int)ei[(size_t)e + wid];
    float nrm = dinv[r] * ew[wid] * dinv[c];
    const float4* xs = (const float4*)(X + (size_t)r * FT) + lane * 2;
    float4 x0 = xs[0], x1 = xs[1];
    float* yd = Y + (size_t)c * FT + lane * 8;
    atomicAdd(yd + 0, nrm * x0.x); atomicAdd(yd + 1, nrm * x0.y);
    atomicAdd(yd + 2, nrm * x0.z); atomicAdd(yd + 3, nrm * x0.w);
    atomicAdd(yd + 4, nrm * x1.x); atomicAdd(yd + 5, nrm * x1.y);
    atomicAdd(yd + 6, nrm * x1.z); atomicAdd(yd + 7, nrm * x1.w);
}

// ---- Phase 3: fold weights, pack WMMA B operands, softmax ------------------
// Packed B layout (contiguous k order chosen here, A matches): lane l holds column
// n = (l&15), halves j=0..15 are K = (l>>4)*16 + j.  B[k][n] = M[n][k].
__global__ void k_prep(const float* __restrict__ Wz, const float* __restrict__ bz,
                       const float* __restrict__ Wh, const float* __restrict__ bh,
                       const float* __restrict__ Lz, const float* __restrict__ lzb,
                       const float* __restrict__ Lh, const float* __restrict__ lhb,
                       const float* __restrict__ att,
                       _Float16* __restrict__ packedB,   // 4 mats * 32 lanes * 16
                       float* __restrict__ cvec,         // cz[32] | ch[32]
                       float* __restrict__ probs) {      // [8]
    __shared__ float Mz[O_CH][F_IN];
    __shared__ float Mh[O_CH][F_IN];
    int tid = threadIdx.x;                                // blockDim = 1024
    {
        int o = tid >> 5, k = tid & 31;
        float sz = 0.f, sh = 0.f;
        #pragma unroll
        for (int j = 0; j < 32; ++j) {
            sz += Lz[o * 64 + j] * Wz[j * 32 + k];
            sh += Lh[o * 64 + j] * Wh[j * 32 + k];
        }
        Mz[o][k] = sz; Mh[o][k] = sh;
    }
    if (tid < 64) {
        int o = tid & 31;
        const float* L  = (tid < 32) ? Lz  : Lh;
        const float* bb = (tid < 32) ? bz  : bh;
        const float* lb = (tid < 32) ? lzb : lhb;
        float s = lb[o];
        #pragma unroll
        for (int j = 0; j < 32; ++j) s += L[o * 64 + j] * bb[j];
        cvec[tid] = s;
    }
    if (tid == 0) {
        float m = att[0];
        for (int t = 1; t < T_STEPS; ++t) m = fmaxf(m, att[t]);
        float ex[T_STEPS], sum = 0.f;
        for (int t = 0; t < T_STEPS; ++t) { ex[t] = __expf(att[t] - m); sum += ex[t]; }
        float inv = 1.0f / sum;
        for (int t = 0; t < T_STEPS; ++t) probs[t] = ex[t] * inv;
    }
    __syncthreads();
    for (int i = tid; i < 4 * 32 * 16; i += blockDim.x) {
        int g = i >> 9;            // 0: Bz(n 0-15) 1: Bz(n 16-31) 2/3: Bh
        int l = (i >> 4) & 31;     // lane
        int j = i & 15;            // half slot
        int n = (l & 15) + ((g & 1) << 4);
        int k = ((l >> 4) << 4) + j;
        const float (*M)[F_IN] = (g < 2) ? Mz : Mh;
        packedB[i] = (_Float16)M[n][k];
    }
}

// Branch-free (1 - sigmoid(z)) * tanh(h) using hardware exp2/rcp.
__device__ __forceinline__ float gate_fuse(float z, float h) {
    float s  = __builtin_amdgcn_rcpf(1.0f + __builtin_amdgcn_exp2f(z * LOG2E));
    float th = 1.0f - 2.0f * __builtin_amdgcn_rcpf(
                   1.0f + __builtin_amdgcn_exp2f(h * (2.0f * LOG2E)));
    return s * th;
}

// ---- Phase 4: gates via WMMA, weighted accumulate --------------------------
// One wave per 16-node tile.  A k order chosen here (same as packed B): lane l holds row
// m=(l&15); halves 0..15 = K (l>>4)*16+0..15.
// D/C layout (f32 16x16): vgpr r, lanes 0-15 -> row r, lanes 16-31 -> row r+8.
__global__ void __launch_bounds__(256)
k_gate(const float* __restrict__ Y,
       const v16h* __restrict__ packedB,
       const float* __restrict__ cvec,
       const float* __restrict__ probs,
       float* __restrict__ out, int n, int nTiles) {
    int wave = (blockIdx.x * blockDim.x + threadIdx.x) >> 5;
    int lane = threadIdx.x & 31;
    if (wave >= nTiles) return;                 // wave-uniform: EXEC stays full
    int m0 = wave * 16;
    int hi = lane >> 4;
    int mr = m0 + (lane & 15);
    if (mr >= n) mr = n - 1;                    // clamp (n % 16 == 0 here)
    const float* yrow = Y + (size_t)mr * FT;

    v16h Bz0 = packedB[0 * 32 + lane];
    v16h Bz1 = packedB[1 * 32 + lane];
    v16h Bh0 = packedB[2 * 32 + lane];
    v16h Bh1 = packedB[3 * 32 + lane];
    float cz0 = cvec[lane & 15], cz1 = cvec[16 + (lane & 15)];
    float ch0 = cvec[32 + (lane & 15)], ch1 = cvec[48 + (lane & 15)];

    v8f acc0 = {}; v8f acc1 = {};
    const v8f zc = {};                          // inline-0 C operand
    for (int t = 0; t < T_STEPS; ++t) {
        v16h a;
        #pragma unroll
        for (int j = 0; j < 8; ++j) {
            a[j]     = (_Float16)yrow[(hi * 16 + j) * T_STEPS + t];
            a[j + 8] = (_Float16)yrow[(hi * 16 + 8 + j) * T_STEPS + t];
        }
        float p = probs[t];

        // n-half 0: only two WMMA results live at a time
        {
            v8f zp = __builtin_amdgcn_wmma_f32_16x16x32_f16(false, a, false, Bz0,
                         (short)0, zc, false, false);
            v8f hp = __builtin_amdgcn_wmma_f32_16x16x32_f16(false, a, false, Bh0,
                         (short)0, zc, false, false);
            #pragma unroll
            for (int r = 0; r < 8; ++r)
                acc0[r] += p * gate_fuse(zp[r] + cz0, hp[r] + ch0);
        }
        // n-half 1
        {
            v8f zp = __builtin_amdgcn_wmma_f32_16x16x32_f16(false, a, false, Bz1,
                         (short)0, zc, false, false);
            v8f hp = __builtin_amdgcn_wmma_f32_16x16x32_f16(false, a, false, Bh1,
                         (short)0, zc, false, false);
            #pragma unroll
            for (int r = 0; r < 8; ++r)
                acc1[r] += p * gate_fuse(zp[r] + cz1, hp[r] + ch1);
        }
    }
    int ncol = lane & 15;
    #pragma unroll
    for (int r = 0; r < 8; ++r) {
        int m = m0 + hi * 8 + r;
        if (m < n) {
            out[(size_t)m * O_CH + ncol]      = acc0[r];
            out[(size_t)m * O_CH + ncol + 16] = acc1[r];
        }
    }
}

// ---------------------------------------------------------------------------
extern "C" void kernel_launch(void* const* d_in, const int* in_sizes, int n_in,
                              void* d_out, int out_size, void* d_ws, size_t ws_size,
                              hipStream_t stream) {
    const float*     X   = (const float*)d_in[0];
    const long long* ei  = (const long long*)d_in[1];
    const float*     ew  = (const float*)d_in[2];
    const float*     Wz  = (const float*)d_in[3];
    const float*     bz  = (const float*)d_in[4];
    // d_in[5], d_in[6] : Wr, br  -- dead (H0 == 0)
    const float*     Wh  = (const float*)d_in[7];
    const float*     bh  = (const float*)d_in[8];
    const float*     Lz  = (const float*)d_in[9];
    const float*     lzb = (const float*)d_in[10];
    // d_in[11], d_in[12] : Lr, lr_b -- dead
    const float*     Lh  = (const float*)d_in[13];
    const float*     lhb = (const float*)d_in[14];
    const float*     att = (const float*)d_in[15];
    float* out = (float*)d_out;

    const int N = in_sizes[0] / FT;
    const int E = in_sizes[2];

    // workspace layout
    float*    deg     = (float*)d_ws;                       // N (-> dinv in place)
    float*    Y       = deg + N;                            // N * 256
    _Float16* packedB = (_Float16*)(Y + (size_t)N * FT);    // 4*32*16 halves
    float*    cvec    = (float*)(packedB + 4 * 32 * 16);    // 64
    float*    probs   = cvec + 64;                          // 8

    k_deg_init <<<(N + 255) / 256, 256, 0, stream>>>(deg, N);
    k_deg_accum<<<(E + 255) / 256, 256, 0, stream>>>(ei, ew, deg, E);
    k_dinv     <<<(N + 255) / 256, 256, 0, stream>>>(deg, N);

    int nvec4 = N * (FT / 4);
    k_y_init   <<<(nvec4 + 255) / 256, 256, 0, stream>>>(
        (const float4*)X, deg, (float4*)Y, N);
    k_edge_prop<<<(E + 7) / 8, 256, 0, stream>>>(X, ei, ew, deg, Y, E);

    k_prep<<<1, 1024, 0, stream>>>(Wz, bz, Wh, bh, Lz, lzb, Lh, lhb, att,
                                   packedB, cvec, probs);

    int nTiles = (N + 15) / 16;
    int nWaveThreads = nTiles * 32;
    k_gate<<<(nWaveThreads + 255) / 256, 256, 0, stream>>>(
        Y, (const v16h*)packedB, cvec, probs, out, N, nTiles);
}
